// AttentionBlock_64776696758780
// MI455X (gfx1250) — matrix operand order repaired and checked by reading, not compiled, not run on hardware
//
#include <hip/hip_runtime.h>
#include <hip/hip_bf16.h>

#define NN 8192
#define DD 256
#define SCALING 0.0625f
#define LOG2E 1.4426950408889634f

typedef __attribute__((ext_vector_type(16))) _Float16 v16h;
typedef __attribute__((ext_vector_type(8)))  _Float16 v8h;
typedef __attribute__((ext_vector_type(8)))  float    v8f;
typedef __attribute__((ext_vector_type(4)))  unsigned int u32x4;
typedef __attribute__((ext_vector_type(8)))  int      i32x8;
typedef __attribute__((ext_vector_type(4)))  int      i32x4;

#define WMMA_F16(a, b, c) \
  __builtin_amdgcn_wmma_f32_16x16x32_f16(false, (a), false, (b), (short)0, (c), false, false)

// ---- Fragment loaders (per CDNA5 ISA §7.12.2 layouts) -----------------------

// A-matrix 16x32 f16: lane -> row (lane&15); halves 0..7 = K b0..b0+7,
// halves 8..15 = K b0+16..b0+23, b0 = (lane<16 ? 0 : 8).
__device__ __forceinline__ v16h load_a16(const _Float16* tile, int ld, int k, int lane) {
  const int m  = lane & 15;
  const int b0 = (lane & 16) ? 8 : 0;
  const _Float16* p = tile + (size_t)m * ld + k + b0;
  v8h lo = *(const v8h*)(p);
  v8h hi = *(const v8h*)(p + 16);
  return __builtin_shufflevector(lo, hi, 0,1,2,3,4,5,6,7,8,9,10,11,12,13,14,15);
}

// B-matrix 32x16 f16: lane -> column (lane&15); halves 0..7 = K b0..b0+7,
// halves 8..15 = K b0+16..b0+23, b0 = (lane<16 ? 0 : 8). Storage is [col][k].
__device__ __forceinline__ v16h load_b16(const _Float16* base, int ld, int col, int k, int lane) {
  const int n  = lane & 15;
  const int kh = (lane & 16) ? 8 : 0;
  const _Float16* p = base + (size_t)(col + n) * ld + k + kh;
  v8h lo = *(const v8h*)(p);
  v8h hi = *(const v8h*)(p + 16);
  return __builtin_shufflevector(lo, hi, 0,1,2,3,4,5,6,7,8,9,10,11,12,13,14,15);
}

// ---- Tensor Data Mover: 2D tile Global -> LDS (ISA 08_async_tensor §8) ------
// D# group0: count=1 | lds_addr | global_addr[56:0] | type=2
// D# group1: workgroup_mask=0, data_size=1 (2B), tensor_dim0/1, tile_dim0/1,
//            tensor_dim0_stride (all in elements). Groups 2/3 unused (2D).
__device__ __forceinline__ void tdm_load_2d(unsigned int lds_off, const void* gaddr,
                                            unsigned int tensor_d0, unsigned int tensor_d1,
                                            unsigned long long stride0,
                                            unsigned int tile_d0, unsigned int tile_d1) {
  const unsigned long long ga = (unsigned long long)(uintptr_t)gaddr;
  u32x4 g0;
  g0.x = 1u;                                                   // count=1, user mode
  g0.y = lds_off;                                              // LDS byte address
  g0.z = (unsigned int)ga;                                     // global_addr[31:0]
  g0.w = (unsigned int)((ga >> 32) & 0x1FFFFFFull) | 0x80000000u; // addr[56:32] | type=2
  i32x8 g1;
  g1[0] = 0x00010000;                                          // data_size=1 (2 bytes)
  g1[1] = (int)((tensor_d0 & 0xFFFFu) << 16);                  // tensor_dim0[15:0]
  g1[2] = (int)((tensor_d0 >> 16) | ((tensor_d1 & 0xFFFFu) << 16));
  g1[3] = (int)((tensor_d1 >> 16) | (tile_d0 << 16));          // tile_dim0
  g1[4] = (int)(tile_d1 & 0xFFFFu);                            // tile_dim1 (tile_dim2=0)
  g1[5] = (int)(unsigned int)stride0;                          // tensor_dim0_stride[31:0]
  g1[6] = (int)((stride0 >> 32) & 0xFFFFull);                  // stride[47:32]
  g1[7] = 0;
  i32x4 z4 = {0, 0, 0, 0};
  i32x8 z8 = {0, 0, 0, 0, 0, 0, 0, 0};
  // 6-arg form (clang-23 / therock-10.0 toolchain)
  __builtin_amdgcn_tensor_load_to_lds(g0, g1, z4, z4, z8, 0);
}

// ---- Kernel 1: x + phi_degree + phi_3d_sum -> f16 ---------------------------

__global__ void prep_x_kernel(const float* __restrict__ x,
                              const float* __restrict__ pdg,
                              const float* __restrict__ p3s,
                              _Float16* __restrict__ xs16) {
  const size_t i = (size_t)blockIdx.x * blockDim.x + threadIdx.x;
  xs16[i] = (_Float16)(x[i] + pdg[i] + p3s[i]);
}

// ---- Kernel 2: transpose + convert weights to f16 ---------------------------

__global__ void prep_w_kernel(const float* __restrict__ Wq,
                              const float* __restrict__ Wk,
                              const float* __restrict__ Wv,
                              _Float16* __restrict__ Wqt,
                              _Float16* __restrict__ Wkt,
                              _Float16* __restrict__ Wvt) {
  const size_t i = (size_t)blockIdx.x * blockDim.x + threadIdx.x;
  const int w   = (int)(i / (DD * DD));
  const int rem = (int)(i % (DD * DD));
  const int k = rem / DD;
  const int n = rem % DD;
  const float* src = (w == 0) ? Wq : (w == 1) ? Wk : Wv;
  _Float16*    dst = (w == 0) ? Wqt : (w == 1) ? Wkt : Wvt;
  dst[(size_t)n * DD + k] = (_Float16)src[(size_t)k * DD + n];
}

// ---- Kernel 3: QKV projections via WMMA -------------------------------------

__global__ __launch_bounds__(32) void qkv_kernel(const _Float16* __restrict__ xs16,
                                                 const _Float16* __restrict__ Wqt,
                                                 const _Float16* __restrict__ Wkt,
                                                 const _Float16* __restrict__ Wvt,
                                                 _Float16* __restrict__ Q16,
                                                 _Float16* __restrict__ K16,
                                                 _Float16* __restrict__ Vt16) {
  const int lane = threadIdx.x & 31;
  const int r0 = blockIdx.x * 16;
  const int c0 = blockIdx.y * 16;
  const int z  = blockIdx.z;
  const _Float16* Wt = (z == 0) ? Wqt : (z == 1) ? Wkt : Wvt;

  v8f acc = {};
#pragma unroll
  for (int t = 0; t < 8; ++t) {
    v16h a = load_a16(xs16 + (size_t)r0 * DD, DD, t * 32, lane);
    v16h b = load_b16(Wt, DD, c0, t * 32, lane);
    acc = WMMA_F16(a, b, acc);
  }

  const int n  = lane & 15;
  const int hi = lane >> 4;
  if (z == 2) {
#pragma unroll
    for (int j = 0; j < 8; ++j)
      Vt16[(size_t)(c0 + n) * NN + r0 + hi * 8 + j] = (_Float16)acc[j];
  } else {
    _Float16* dst = (z == 0) ? Q16 : K16;
#pragma unroll
    for (int j = 0; j < 8; ++j)
      dst[(size_t)(r0 + hi * 8 + j) * DD + c0 + n] = (_Float16)acc[j];
  }
}

// ---- Kernel 4: flash attention, TDM-fed LDS tiles, fused bias add -----------
// 4 waves/block, each wave owns 16 query rows (O accumulator 16x256 f32 in
// VGPRs). K/V tiles (32 keys) are DMA'd into double-buffered LDS by the TDM
// once per block instead of per wave (4x less L2 traffic). Bias matrices are
// streamed once from HBM with non-temporal hints.

__global__ __launch_bounds__(128, 1) void attn_kernel(const _Float16* __restrict__ Q16,
                                                      const _Float16* __restrict__ K16,
                                                      const _Float16* __restrict__ Vt16,
                                                      const float* __restrict__ pspd,
                                                      const float* __restrict__ pedg,
                                                      const float* __restrict__ p3d,
                                                      float* __restrict__ out) {
  const int lane = threadIdx.x & 31;
  const int wave = threadIdx.x >> 5;
  const int r0 = (blockIdx.x * 4 + wave) * 16;
  const int n  = lane & 15;
  const int hi = lane >> 4;

  __shared__ __align__(16) _Float16 Kbuf[2][32 * DD];   // [key][d]   16KB x2
  __shared__ __align__(16) _Float16 Vbuf[2][DD * 32];   // [dcol][key] 16KB x2
  __shared__ __align__(16) _Float16 pstage[4][16 * 32]; // per-wave P staging
  _Float16* myP = pstage[wave];

  // Q fragments for full D=256, resident in registers (8 x v16h)
  v16h aQ[8];
#pragma unroll
  for (int t = 0; t < 8; ++t)
    aQ[t] = load_a16(Q16 + (size_t)r0 * DD, DD, t * 32, lane);

  float mrow[8], lsum[8];
  v8f   O[16];
#pragma unroll
  for (int j = 0; j < 8; ++j) { mrow[j] = -__builtin_inff(); lsum[j] = 0.0f; }
#pragma unroll
  for (int t = 0; t < 16; ++t) O[t] = (v8f){};

  // prologue: DMA first K/V tiles into buffer 0
  if (wave == 0) {
    tdm_load_2d((unsigned int)(uintptr_t)&Kbuf[0][0], K16, DD, NN, DD, DD, 32);
    tdm_load_2d((unsigned int)(uintptr_t)&Vbuf[0][0], Vt16, NN, DD, NN, 32, DD);
  }

  int ib = 0;
  for (int k0 = 0; k0 < NN; k0 += 32, ib ^= 1) {
    if (wave == 0) {
      if (k0 + 32 < NN) {
        // issue next tiles into the other buffer, then wait for current pair
        tdm_load_2d((unsigned int)(uintptr_t)&Kbuf[ib ^ 1][0],
                    K16 + (size_t)(k0 + 32) * DD, DD, NN, DD, DD, 32);
        tdm_load_2d((unsigned int)(uintptr_t)&Vbuf[ib ^ 1][0],
                    Vt16 + (k0 + 32), NN, DD, NN, 32, DD);
        __builtin_amdgcn_s_wait_tensorcnt(2);
      } else {
        __builtin_amdgcn_s_wait_tensorcnt(0);
      }
    }
    __syncthreads();

    const _Float16* Kb = Kbuf[ib];
    const _Float16* Vb = Vbuf[ib];

    // ---- S = Q K^T for 32 keys (two 16x16 tiles), 16 WMMAs, B from LDS ----
    v8f S0 = {}, S1 = {};
#pragma unroll
    for (int t = 0; t < 8; ++t) {
      v16h b0 = load_b16(Kb, DD, 0,  t * 32, lane);
      v16h b1 = load_b16(Kb, DD, 16, t * 32, lane);
      S0 = WMMA_F16(aQ[t], b0, S0);
      S1 = WMMA_F16(aQ[t], b1, S1);
    }

    // ---- bias add (non-temporal, read-once HBM stream) + online softmax ----
    const int key0 = k0 + n;
#pragma unroll
    for (int j = 0; j < 8; ++j) {
      const size_t rowoff = (size_t)(r0 + hi * 8 + j) * NN;
      float s0 = S0[j] * SCALING
               + __builtin_nontemporal_load(pspd + rowoff + key0)
               + __builtin_nontemporal_load(pedg + rowoff + key0)
               + __builtin_nontemporal_load(p3d  + rowoff + key0);
      float s1 = S1[j] * SCALING
               + __builtin_nontemporal_load(pspd + rowoff + key0 + 16)
               + __builtin_nontemporal_load(pedg + rowoff + key0 + 16)
               + __builtin_nontemporal_load(p3d  + rowoff + key0 + 16);

      float tm = fmaxf(s0, s1);
      tm = fmaxf(tm, __shfl_xor(tm, 1, 16));
      tm = fmaxf(tm, __shfl_xor(tm, 2, 16));
      tm = fmaxf(tm, __shfl_xor(tm, 4, 16));
      tm = fmaxf(tm, __shfl_xor(tm, 8, 16));

      const float mnew  = fmaxf(mrow[j], tm);
      const float scale = exp2f((mrow[j] - mnew) * LOG2E);
      const float p0 = exp2f((s0 - mnew) * LOG2E);
      const float p1 = exp2f((s1 - mnew) * LOG2E);

      float ps = p0 + p1;
      ps += __shfl_xor(ps, 1, 16);
      ps += __shfl_xor(ps, 2, 16);
      ps += __shfl_xor(ps, 4, 16);
      ps += __shfl_xor(ps, 8, 16);

      lsum[j] = lsum[j] * scale + ps;
      mrow[j] = mnew;

#pragma unroll
      for (int t = 0; t < 16; ++t) O[t][j] *= scale;

      // stage P (C-layout -> row-major in LDS) for A-fragment reload
      myP[(hi * 8 + j) * 32 + n]      = (_Float16)p0;
      myP[(hi * 8 + j) * 32 + 16 + n] = (_Float16)p1;
    }

    // wave-lockstep LDS transpose: wait for our own DS stores, then reload
    asm volatile("s_wait_dscnt 0" ::: "memory");

    v16h aP;
    {
      const int m  = lane & 15;
      const int b0 = (lane & 16) ? 8 : 0;
      const _Float16* p = myP + m * 32 + b0;
      v8h lo = *(const v8h*)(p);
      v8h hip_ = *(const v8h*)(p + 16);
      aP = __builtin_shufflevector(lo, hip_, 0,1,2,3,4,5,6,7,8,9,10,11,12,13,14,15);
    }

    // ---- O += P V for all 256 output columns, 16 WMMAs, B from LDS ----
#pragma unroll
    for (int t = 0; t < 16; ++t) {
      v16h bv = load_b16(Vb, 32, t * 16, 0, lane);
      O[t] = WMMA_F16(aP, bv, O[t]);
    }

    __syncthreads();
  }

  // ---- finalize: divide by softmax denominator, non-temporal store ----
#pragma unroll
  for (int j = 0; j < 8; ++j) lsum[j] = 1.0f / lsum[j];
#pragma unroll
  for (int t = 0; t < 16; ++t) {
#pragma unroll
    for (int j = 0; j < 8; ++j)
      __builtin_nontemporal_store(O[t][j] * lsum[j],
                                  out + (size_t)(r0 + hi * 8 + j) * DD + t * 16 + n);
  }
}

// ---- host launch ------------------------------------------------------------

extern "C" void kernel_launch(void* const* d_in, const int* in_sizes, int n_in,
                              void* d_out, int out_size, void* d_ws, size_t ws_size,
                              hipStream_t stream) {
  const float* x   = (const float*)d_in[0];
  const float* pdg = (const float*)d_in[1];
  const float* p3s = (const float*)d_in[2];
  const float* p3d = (const float*)d_in[3];
  const float* psp = (const float*)d_in[4];
  const float* ped = (const float*)d_in[5];
  const float* Wq  = (const float*)d_in[6];
  const float* Wk  = (const float*)d_in[7];
  const float* Wv  = (const float*)d_in[8];
  float* out = (float*)d_out;

  _Float16* xs16 = (_Float16*)d_ws;
  _Float16* Wqt  = xs16 + (size_t)NN * DD;
  _Float16* Wkt  = Wqt + (size_t)DD * DD;
  _Float16* Wvt  = Wkt + (size_t)DD * DD;
  _Float16* Q16  = Wvt + (size_t)DD * DD;
  _Float16* K16  = Q16 + (size_t)NN * DD;
  _Float16* Vt16 = K16 + (size_t)NN * DD;

  prep_x_kernel<<<(NN * DD) / 256, 256, 0, stream>>>(x, pdg, p3s, xs16);
  prep_w_kernel<<<(3 * DD * DD) / 256, 256, 0, stream>>>(Wq, Wk, Wv, Wqt, Wkt, Wvt);
  qkv_kernel<<<dim3(NN / 16, DD / 16, 3), 32, 0, stream>>>(xs16, Wqt, Wkt, Wvt, Q16, K16, Vt16);
  attn_kernel<<<NN / 64, 128, 0, stream>>>(Q16, K16, Vt16, psp, ped, p3d, out);
}
